// StandardMultiHeadAttention_27393301413963
// MI455X (gfx1250) — matrix operand order repaired and checked by reading, not compiled, not run on hardware
//
#include <hip/hip_runtime.h>
#include <hip/hip_bf16.h>

// ---------------------------------------------------------------------------
// Causal MHA for MI455X (gfx1250, wave32):
//   stage 0: fp32 -> bf16 (+ weight transpose so B-fragments are contiguous)
//   stage 1: qkv = x @ Wqkv -> Q(prescaled), K, V^T    (32x64 tile/wave, WMMA)
//   stage 2: flash attention, 16-query tile per wave   (QK^T + ones-sum + PV WMMA)
//   stage 3: out = attn @ Wout -> fp32                 (32x64 tile/wave, WMMA)
// ---------------------------------------------------------------------------

typedef __attribute__((ext_vector_type(16))) __bf16 v16bf;
typedef __attribute__((ext_vector_type(8)))  __bf16 v8bf;
typedef __attribute__((ext_vector_type(8)))  float  v8f;

#define D_MODEL 1024
#define NHEADS  16
#define HDIM    64
#define BATCH   2
#define SEQ     2048
#define MROWS   (BATCH * SEQ)   // 4096
#define NQKV    (3 * D_MODEL)   // 3072

// ---- WMMA fragment helpers (16x16x32 bf16, contiguous k order used by this file) -------------------
// A (16x32, MxK): lane L holds row M=L%16; elems i -> K = k0 + (L/16)*16 + i
static __device__ inline v16bf load_fragA(const __bf16* rowptr, int k0, int half) {
  return *(const v16bf*)(rowptr + k0 + half * 16);
}

// B (32x16, KxN): lane L holds column N=L%16; elems i -> K = k0 + (L/16)*16 + i
// (one contiguous 32B load per lane; colptr must be contiguous in K)
static __device__ inline v16bf load_fragB(const __bf16* colptr, int k0, int half) {
  return *(const v16bf*)(colptr + k0 + half * 16);
}

static __device__ inline v8f wmma_bf16(v16bf a, v16bf b, v8f c) {
  return __builtin_amdgcn_wmma_f32_16x16x32_bf16(false, a, false, b,
                                                 (short)0, c, false, false);
}

// ---- cross-lane max within 16-lane halves (wave32, ds_swizzle) -------------
template <int XMASK>
static __device__ inline float xor_bcast(float v) {
  int i = __builtin_bit_cast(int, v);
  i = __builtin_amdgcn_ds_swizzle(i, (XMASK << 10) | 0x1f);  // group-of-32 xor
  return __builtin_bit_cast(float, i);
}
static __device__ inline float redmax16(float v) {
  v = fmaxf(v, xor_bcast<1>(v)); v = fmaxf(v, xor_bcast<2>(v));
  v = fmaxf(v, xor_bcast<4>(v)); v = fmaxf(v, xor_bcast<8>(v));
  return v;
}

// ---- stage 0: fp32 -> bf16 conversions -------------------------------------
__global__ void cvt_bf16_kernel(const float* __restrict__ src,
                                __bf16* __restrict__ dst, int n) {
  int i = blockIdx.x * blockDim.x + threadIdx.x;
  if (i < n) dst[i] = (__bf16)src[i];
}

// src [rows][cols] fp32 -> dst [cols][rows] bf16 (transpose for B-fragments)
__global__ void cvt_transpose_kernel(const float* __restrict__ src,
                                     __bf16* __restrict__ dst,
                                     int rows, int cols) {
  int i = blockIdx.x * blockDim.x + threadIdx.x;
  if (i < rows * cols) {
    int r = i / cols, c = i % cols;
    dst[c * rows + r] = (__bf16)src[i];
  }
}

// ---- stage 1: QKV projection, 32x64 tile per wave --------------------------
// xb [4096][1024] bf16, wT [3072][1024] bf16 (transposed Wqkv).
// Writes: Q [B,H,L,Dh] (pre-scaled 1/sqrt(Dh)), K [B,H,L,Dh], VT [B,H,Dh,L].
__global__ void qkv_gemm_kernel(const __bf16* __restrict__ xb,
                                const __bf16* __restrict__ wT,
                                __bf16* __restrict__ Q,
                                __bf16* __restrict__ K,
                                __bf16* __restrict__ VT) {
  const int wave = threadIdx.x >> 5;
  const int lane = threadIdx.x & 31;
  const int l16  = lane & 15;
  const int half = lane >> 4;
  const int ntn  = NQKV / 64;                      // 48 column tiles
  int tile  = blockIdx.x * (blockDim.x >> 5) + wave;
  int mbase = (tile / ntn) * 32;
  int nbase = (tile % ntn) * 64;

  const __bf16* arow0 = xb + (size_t)(mbase + l16) * D_MODEL;
  const __bf16* arow1 = arow0 + (size_t)16 * D_MODEL;
  const __bf16* brow  = wT + (size_t)(nbase + l16) * D_MODEL;

  v8f acc[2][4] = {};
#pragma unroll 2
  for (int k0 = 0; k0 < D_MODEL; k0 += 32) {
    __builtin_prefetch(arow0 + k0 + 512, 0, 0);
    __builtin_prefetch(brow + k0 + 512, 0, 0);
    v16bf a0 = load_fragA(arow0, k0, half);
    v16bf a1 = load_fragA(arow1, k0, half);
    v16bf b0 = load_fragB(brow + (size_t)(0 * 16) * D_MODEL, k0, half);
    v16bf b1 = load_fragB(brow + (size_t)(1 * 16) * D_MODEL, k0, half);
    v16bf b2 = load_fragB(brow + (size_t)(2 * 16) * D_MODEL, k0, half);
    v16bf b3 = load_fragB(brow + (size_t)(3 * 16) * D_MODEL, k0, half);
    acc[0][0] = wmma_bf16(a0, b0, acc[0][0]);
    acc[0][1] = wmma_bf16(a0, b1, acc[0][1]);
    acc[0][2] = wmma_bf16(a0, b2, acc[0][2]);
    acc[0][3] = wmma_bf16(a0, b3, acc[0][3]);
    acc[1][0] = wmma_bf16(a1, b0, acc[1][0]);
    acc[1][1] = wmma_bf16(a1, b1, acc[1][1]);
    acc[1][2] = wmma_bf16(a1, b2, acc[1][2]);
    acc[1][3] = wmma_bf16(a1, b3, acc[1][3]);
  }

  // 64-wide N tile sits inside exactly one of {Q,K,V} and one head.
  const int which = nbase >> 10;                   // 0=Q 1=K 2=V (uniform)
  const int h     = (nbase & (D_MODEL - 1)) >> 6;  // head (uniform)
#pragma unroll
  for (int rb = 0; rb < 2; ++rb) {
#pragma unroll
    for (int r = 0; r < 8; ++r) {
      int m  = mbase + rb * 16 + r + 8 * half;
      int bb = m >> 11, l = m & (SEQ - 1);
      int hq = bb * NHEADS + h;
#pragma unroll
      for (int nc = 0; nc < 4; ++nc) {
        int d = nc * 16 + l16;
        float v = acc[rb][nc][r];
        if (which == 0) {
          Q[((size_t)hq * SEQ + l) * HDIM + d] = (__bf16)(v * 0.125f);
        } else if (which == 1) {
          K[((size_t)hq * SEQ + l) * HDIM + d] = (__bf16)v;
        } else {
          VT[((size_t)hq * HDIM + d) * SEQ + l] = (__bf16)v;
        }
      }
    }
  }
}

// ---- stage 2: causal flash attention, one 16-row query tile per wave -------
#define ATT_WAVES 4
__global__ void attn_kernel(const __bf16* __restrict__ Q,
                            const __bf16* __restrict__ K,
                            const __bf16* __restrict__ VT,
                            __bf16* __restrict__ AO) {
  __shared__ __attribute__((aligned(64))) __bf16 Plds[ATT_WAVES][16][32];
  const int wave = threadIdx.x >> 5;
  const int lane = threadIdx.x & 31;
  const int l16  = lane & 15;
  const int half = lane >> 4;

  int tile  = blockIdx.x * ATT_WAVES + wave;
  int qt    = tile & 127;                 // 128 q-tiles per head
  int h     = (tile >> 7) & (NHEADS - 1);
  int bb    = tile >> 11;
  int qbase = qt * 16;

  const __bf16* qrow = Q + (size_t)(((bb * NHEADS + h) * SEQ) + qbase + l16) * HDIM;
  v16bf aq0 = load_fragA(qrow, 0, half);
  v16bf aq1 = load_fragA(qrow, 32, half);

  v16bf onesb;
#pragma unroll
  for (int i = 0; i < 16; ++i) onesb[i] = (__bf16)1.0f;

  v8f o0 = {}, o1 = {}, o2 = {}, o3 = {};
  float mrow[8], lrow[8];
#pragma unroll
  for (int r = 0; r < 8; ++r) { mrow[r] = -1e30f; lrow[r] = 0.f; }

  const __bf16* kbaseptr = K  + (size_t)((bb * NHEADS + h) * SEQ) * HDIM;
  const __bf16* vbaseptr = VT + (size_t)((bb * NHEADS + h) * HDIM) * SEQ;
  const int ntile = ((qbase + 15) >> 5) + 1;        // causal: 32-wide key tiles

  for (int kt = 0; kt < ntile; ++kt) {
    int kb = kt * 32;
    const __bf16* krow0 = kbaseptr + (size_t)(kb + l16) * HDIM;
    const __bf16* krow1 = krow0 + 16 * HDIM;
    v8f s0 = {}, s1 = {};
    s0 = wmma_bf16(aq0, load_fragB(krow0, 0, half), s0);
    s0 = wmma_bf16(aq1, load_fragB(krow0, 32, half), s0);
    s1 = wmma_bf16(aq0, load_fragB(krow1, 0, half), s1);
    s1 = wmma_bf16(aq1, load_fragB(krow1, 32, half), s1);

    int col0 = kb + l16, col1 = col0 + 16;
#pragma unroll
    for (int r = 0; r < 8; ++r) {
      int row  = qbase + r + 8 * half;
      float v0 = (col0 > row) ? -1e30f : s0[r];
      float v1 = (col1 > row) ? -1e30f : s1[r];
      float tm   = redmax16(fmaxf(v0, v1));
      float mnew = fmaxf(mrow[r], tm);
      float alpha = __expf(mrow[r] - mnew);
      mrow[r] = mnew;
      lrow[r] *= alpha;
      o0[r] *= alpha; o1[r] *= alpha; o2[r] *= alpha; o3[r] *= alpha;
      // C-layout -> A-layout transpose of P through LDS
      Plds[wave][r + 8 * half][l16]      = (__bf16)__expf(v0 - mnew);
      Plds[wave][r + 8 * half][16 + l16] = (__bf16)__expf(v1 - mnew);
    }
    asm volatile("s_wait_dscnt 0" ::: "memory");
    v16bf ap = load_fragA(&Plds[wave][l16][0], 0, half);

    // row sums of P via WMMA against all-ones B (replaces 32 swizzle-adds)
    v8f sacc = {};
    sacc = wmma_bf16(ap, onesb, sacc);

    const __bf16* vr0 = vbaseptr + (size_t)l16 * SEQ;
    o0 = wmma_bf16(ap, load_fragB(vr0,            kb, half), o0);
    o1 = wmma_bf16(ap, load_fragB(vr0 + 16 * SEQ, kb, half), o1);
    o2 = wmma_bf16(ap, load_fragB(vr0 + 32 * SEQ, kb, half), o2);
    o3 = wmma_bf16(ap, load_fragB(vr0 + 48 * SEQ, kb, half), o3);
#pragma unroll
    for (int r = 0; r < 8; ++r) lrow[r] += sacc[r];
  }

#pragma unroll
  for (int r = 0; r < 8; ++r) {
    float inv = 1.0f / lrow[r];
    int row = qbase + r + 8 * half;
    __bf16* dst = AO + ((size_t)bb * SEQ + row) * D_MODEL + h * HDIM;
    dst[l16]      = (__bf16)(o0[r] * inv);
    dst[16 + l16] = (__bf16)(o1[r] * inv);
    dst[32 + l16] = (__bf16)(o2[r] * inv);
    dst[48 + l16] = (__bf16)(o3[r] * inv);
  }
}

// ---- stage 3: output projection -> fp32, 32x64 tile per wave ---------------
__global__ void out_gemm_kernel(const __bf16* __restrict__ AO,
                                const __bf16* __restrict__ wT,
                                float* __restrict__ out) {
  const int wave = threadIdx.x >> 5;
  const int lane = threadIdx.x & 31;
  const int l16  = lane & 15;
  const int half = lane >> 4;
  const int ntn  = D_MODEL / 64;                   // 16 column tiles
  int tile  = blockIdx.x * (blockDim.x >> 5) + wave;
  int mbase = (tile / ntn) * 32;
  int nbase = (tile % ntn) * 64;

  const __bf16* arow0 = AO + (size_t)(mbase + l16) * D_MODEL;
  const __bf16* arow1 = arow0 + (size_t)16 * D_MODEL;
  const __bf16* brow  = wT + (size_t)(nbase + l16) * D_MODEL;

  v8f acc[2][4] = {};
#pragma unroll 2
  for (int k0 = 0; k0 < D_MODEL; k0 += 32) {
    __builtin_prefetch(arow0 + k0 + 512, 0, 0);
    v16bf a0 = load_fragA(arow0, k0, half);
    v16bf a1 = load_fragA(arow1, k0, half);
    v16bf b0 = load_fragB(brow + (size_t)(0 * 16) * D_MODEL, k0, half);
    v16bf b1 = load_fragB(brow + (size_t)(1 * 16) * D_MODEL, k0, half);
    v16bf b2 = load_fragB(brow + (size_t)(2 * 16) * D_MODEL, k0, half);
    v16bf b3 = load_fragB(brow + (size_t)(3 * 16) * D_MODEL, k0, half);
    acc[0][0] = wmma_bf16(a0, b0, acc[0][0]);
    acc[0][1] = wmma_bf16(a0, b1, acc[0][1]);
    acc[0][2] = wmma_bf16(a0, b2, acc[0][2]);
    acc[0][3] = wmma_bf16(a0, b3, acc[0][3]);
    acc[1][0] = wmma_bf16(a1, b0, acc[1][0]);
    acc[1][1] = wmma_bf16(a1, b1, acc[1][1]);
    acc[1][2] = wmma_bf16(a1, b2, acc[1][2]);
    acc[1][3] = wmma_bf16(a1, b3, acc[1][3]);
  }
#pragma unroll
  for (int rb = 0; rb < 2; ++rb) {
#pragma unroll
    for (int r = 0; r < 8; ++r) {
      int m = mbase + rb * 16 + r + 8 * half;
#pragma unroll
      for (int nc = 0; nc < 4; ++nc) {
        out[(size_t)m * D_MODEL + nbase + nc * 16 + l16] = acc[rb][nc][r];
      }
    }
  }
}

// ---------------------------------------------------------------------------
extern "C" void kernel_launch(void* const* d_in, const int* in_sizes, int n_in,
                              void* d_out, int out_size, void* d_ws, size_t ws_size,
                              hipStream_t stream) {
  (void)in_sizes; (void)n_in; (void)out_size; (void)ws_size;
  const float* x     = (const float*)d_in[0];
  const float* w_qkv = (const float*)d_in[1];
  const float* w_out = (const float*)d_in[2];
  float* out = (float*)d_out;

  char* ws = (char*)d_ws;
  const size_t MB = 1024 * 1024;
  __bf16* xb    = (__bf16*)(ws);                  //  8 MB: x bf16 [4096][1024]
  __bf16* wqkvT = (__bf16*)(ws + 8 * MB);         //  6 MB: Wqkv^T [3072][1024]
  __bf16* woutT = (__bf16*)(ws + 14 * MB);        //  2 MB: Wout^T [1024][1024]
  __bf16* Q     = (__bf16*)(ws + 16 * MB);        //  8 MB: [B,H,L,Dh]
  __bf16* K     = (__bf16*)(ws + 24 * MB);        //  8 MB: [B,H,L,Dh]
  __bf16* VT    = (__bf16*)(ws + 32 * MB);        //  8 MB: [B,H,Dh,L]
  __bf16* AO    = (__bf16*)(ws + 40 * MB);        //  8 MB: attn out [B,L,D]

  int nx = MROWS * D_MODEL;
  cvt_bf16_kernel<<<(nx + 255) / 256, 256, 0, stream>>>(x, xb, nx);
  cvt_transpose_kernel<<<(D_MODEL * NQKV + 255) / 256, 256, 0, stream>>>(
      w_qkv, wqkvT, D_MODEL, NQKV);
  cvt_transpose_kernel<<<(D_MODEL * D_MODEL + 255) / 256, 256, 0, stream>>>(
      w_out, woutT, D_MODEL, D_MODEL);

  {
    int tiles = (MROWS / 32) * (NQKV / 64);        // 6144 waves
    qkv_gemm_kernel<<<tiles / 8, 256, 0, stream>>>(xb, wqkvT, Q, K, VT);
  }
  {
    int tiles = BATCH * NHEADS * (SEQ / 16);       // 4096 waves
    attn_kernel<<<tiles / ATT_WAVES, 32 * ATT_WAVES, 0, stream>>>(Q, K, VT, AO);
  }
  {
    int tiles = (MROWS / 32) * (D_MODEL / 64);     // 2048 waves
    out_gemm_kernel<<<tiles / 8, 256, 0, stream>>>(AO, woutT, out);
  }
}
